// MiniMaxText01LinearAttention_84370337563031
// MI455X (gfx1250) — compile-verified, matrix operand order checked by reading, not run on hardware
//
#include <hip/hip_runtime.h>

// MiniMaxText01 lightning attention for gfx1250 (MI455X), bf16 WMMA pipeline.
// v3: double-buffered async global->LDS software pipeline in the GEMMs
// (issue tile k+1, compute tile k on WMMA, then s_wait_asynccnt + barrier).

#define USE_ASYNC_LDS 1

typedef __attribute__((ext_vector_type(16))) __bf16 v16bf;
typedef __attribute__((ext_vector_type(8)))  float  v8f;

#define BB   2
#define SS   4096
#define HIDD 2048
#define HH   16
#define DD   128
#define CC   256
#define NCC  16

__device__ __forceinline__ unsigned short f2bf(float f) {
  union { __bf16 h; unsigned short s; } u;
  u.h = (__bf16)f;                       // hardware RNE cvt
  return u.s;
}
__device__ __forceinline__ float bf2f(unsigned short v) {
  union { unsigned short s; __bf16 h; } u;
  u.s = v;
  return (float)u.h;
}
__device__ __forceinline__ v8f vzero8() {
  v8f z = {0.f, 0.f, 0.f, 0.f, 0.f, 0.f, 0.f, 0.f};
  return z;
}
__device__ __forceinline__ v8f wmma_bf16(v16bf a, v16bf b, v8f c) {
  return __builtin_amdgcn_wmma_f32_16x16x32_bf16(false, a, false, b, (short)0, c,
                                                 false, false);
}

// Async copy of one 32-byte row segment (both halves) into LDS; the inst
// offset applies to BOTH the LDS and global address (ISA 08 §4.4).
__device__ __forceinline__ void async_row32(unsigned ldsoff,
                                            const unsigned short* g) {
  asm volatile(
      "global_load_async_to_lds_b128 %0, %1, off\n\t"
      "global_load_async_to_lds_b128 %0, %1, off offset:16"
      :
      : "v"(ldsoff), "v"(g)
      : "memory");
}
__device__ __forceinline__ void wait_async0() {
  asm volatile("s_wait_asynccnt 0" ::: "memory");
}

// A/B 16x32 bf16 fragment per ISA layout (two aligned b128 LDS reads).
__device__ __forceinline__ v16bf frag_ld(const unsigned short* p, int stride,
                                         int row, int k0) {
  int lane = threadIdx.x & 31;
  int r = row + (lane & 15);
  int kh = k0 + ((lane >> 4) << 3);
  union { uint4 u[2]; v16bf v; } f;
  f.u[0] = *(const uint4*)(p + r * stride + kh);
  f.u[1] = *(const uint4*)(p + r * stride + kh + 16);
  return f.v;
}
__device__ __forceinline__ v16bf frag_ld_g(const unsigned short* p, int k0) {
  int lane = threadIdx.x & 31;
  int r = lane & 15;
  int kh = k0 + ((lane >> 4) << 3);
  union { uint4 u[2]; v16bf v; } f;
  f.u[0] = *(const uint4*)(p + (size_t)r * 128 + kh);
  f.u[1] = *(const uint4*)(p + (size_t)r * 128 + kh + 16);
  return f.v;
}
// B fragment from the f32 kv state (stored [e][d], stride 132 f32), cvt->bf16.
__device__ __forceinline__ v16bf frag_ld_kv(const float* p, int nbase, int k0) {
  int lane = threadIdx.x & 31;
  int n = nbase + (lane & 15);
  int kh = k0 + ((lane >> 4) << 3);
  union { unsigned short sh[16]; v16bf v; } f;
#pragma unroll
  for (int u = 0; u < 8; ++u) {
    f.sh[u]     = f2bf(p[n * 132 + kh + u]);
    f.sh[u + 8] = f2bf(p[n * 132 + kh + 16 + u]);
  }
  return f.v;
}
// A fragment of k^T (transposed gather from k_lds[j][d]) with key decay
// kdec(j) = dtab[C-1-j] fused in.
__device__ __forceinline__ v16bf frag_ld_kt(const unsigned short* p,
                                            const float* dtab, int dbase,
                                            int j0) {
  int lane = threadIdx.x & 31;
  int d = dbase + (lane & 15);
  int jh = j0 + ((lane >> 4) << 3);
  union { unsigned short sh[16]; v16bf v; } f;
#pragma unroll
  for (int u = 0; u < 8; ++u) {
    int j1 = jh + u, j2 = jh + 16 + u;
    f.sh[u]     = f2bf(bf2f(p[j1 * 136 + d]) * dtab[CC - 1 - j1]);
    f.sh[u + 8] = f2bf(bf2f(p[j2 * 136 + d]) * dtab[CC - 1 - j2]);
  }
  return f.v;
}
__device__ __forceinline__ v16bf frag_scale(v16bf a, float sc) {
  union { v16bf v; unsigned short s[16]; } f; f.v = a;
#pragma unroll
  for (int i = 0; i < 16; ++i) f.s[i] = f2bf(bf2f(f.s[i]) * sc);
  return f.v;
}

// ---- one-shot conversion kernels -------------------------------------------
__global__ __launch_bounds__(256) void cvt_bf16(const float* __restrict__ s,
                                                unsigned short* __restrict__ d) {
  size_t i = ((size_t)blockIdx.x * 256 + threadIdx.x) * 4;
  float4 f = *(const float4*)(s + i);
  union { unsigned short s4[4]; uint2 u; } o;
  o.s4[0] = f2bf(f.x); o.s4[1] = f2bf(f.y);
  o.s4[2] = f2bf(f.z); o.s4[3] = f2bf(f.w);
  *(uint2*)(d + i) = o.u;
}
// w[K][N] f32 -> wT[N][K] bf16, 32x32 LDS tile transpose
__global__ __launch_bounds__(256) void cvt_wT(const float* __restrict__ w,
                                              unsigned short* __restrict__ wt,
                                              int K, int N) {
  __shared__ float t[32][33];
  int kt = blockIdx.y * 32, nt = blockIdx.x * 32;
  int tx = threadIdx.x & 31, ty = threadIdx.x >> 5;
#pragma unroll
  for (int r = 0; r < 4; ++r)
    t[ty + 8 * r][tx] = w[(size_t)(kt + ty + 8 * r) * N + nt + tx];
  __syncthreads();
#pragma unroll
  for (int r = 0; r < 4; ++r)
    wt[(size_t)(nt + ty + 8 * r) * K + kt + tx] = f2bf(t[tx][ty + 8 * r]);
}

// ---- GEMM: C[M,N] = act(A[M,K] @ B[K,N]); A bf16 [M][K], B^T bf16 [N][K] ----
// Double-buffered async pipeline: stage tile k+1 while WMMAs run on tile k.
// MODE 0: silu, scatter q/k/v bf16 chunked [B,H,S,D]
// MODE 1: sigmoid -> gate bf16 [B*S, H*D]
// MODE 2: plain f32 store -> d_out
template <int MODE>
__global__ __launch_bounds__(256) void gemm_k(
    const unsigned short* __restrict__ Ah, const unsigned short* __restrict__ Bt,
    int M, int N, int K,
    unsigned short* __restrict__ qb, unsigned short* __restrict__ kb,
    unsigned short* __restrict__ vb, unsigned short* __restrict__ gb,
    float* __restrict__ of) {
  __shared__ unsigned short lA[2][128 * 40];   // [m][k], stride 40
  __shared__ unsigned short lB[2][128 * 40];   // [n][k], stride 40

  const int tid = threadIdx.x;
  const int lane = tid & 31;
  const int wid = tid >> 5;
  const int wm = (wid & 1) * 64;
  const int wn = (wid >> 1) * 32;
  const int bm = blockIdx.y * 128;
  const int bn = blockIdx.x * 128;
  const int sr = tid >> 1;             // staging row 0..127 (2 threads/row)
  const int sc = (tid & 1) * 16;       // staging col base {0,16}

  v8f acc[4][2];
#pragma unroll
  for (int i = 0; i < 4; ++i)
#pragma unroll
    for (int j = 0; j < 2; ++j) acc[i][j] = vzero8();

  const unsigned short* gA = Ah + (size_t)(bm + sr) * K + sc;
  const unsigned short* gB = Bt + (size_t)(bn + sr) * K + sc;
  const unsigned la0 = (unsigned)(unsigned long long)&lA[0][sr * 40 + sc];
  const unsigned la1 = (unsigned)(unsigned long long)&lA[1][sr * 40 + sc];
  const unsigned lb0 = (unsigned)(unsigned long long)&lB[0][sr * 40 + sc];
  const unsigned lb1 = (unsigned)(unsigned long long)&lB[1][sr * 40 + sc];

#if USE_ASYNC_LDS
  // prologue: tile 0 -> buffer 0
  async_row32(la0, gA);
  async_row32(lb0, gB);
  wait_async0();
  __syncthreads();
#endif

  for (int k0 = 0; k0 < K; k0 += 64) {
    // ---- phase 0: compute buf0 (tile k0), async-stage tile k0+32 -> buf1
#if USE_ASYNC_LDS
    async_row32(la1, gA + k0 + 32);
    async_row32(lb1, gB + k0 + 32);
#else
    {
      uint4 a0 = *(const uint4*)(gA + k0);
      uint4 a1 = *(const uint4*)(gA + k0 + 8);
      uint4 b0 = *(const uint4*)(gB + k0);
      uint4 b1 = *(const uint4*)(gB + k0 + 8);
      __syncthreads();
      *(uint4*)&lA[0][sr * 40 + sc] = a0;
      *(uint4*)&lA[0][sr * 40 + sc + 8] = a1;
      *(uint4*)&lB[0][sr * 40 + sc] = b0;
      *(uint4*)&lB[0][sr * 40 + sc + 8] = b1;
      __syncthreads();
    }
#endif
    {
      v16bf bf0 = frag_ld(lB[0], 40, wn, 0);
      v16bf bf1 = frag_ld(lB[0], 40, wn + 16, 0);
#pragma unroll
      for (int mt = 0; mt < 4; ++mt) {
        v16bf a = frag_ld(lA[0], 40, wm + mt * 16, 0);
        acc[mt][0] = wmma_bf16(a, bf0, acc[mt][0]);
        acc[mt][1] = wmma_bf16(a, bf1, acc[mt][1]);
      }
    }
#if USE_ASYNC_LDS
    wait_async0();
    __syncthreads();
#endif
    // ---- phase 1: compute buf1 (tile k0+32), async-stage tile k0+64 -> buf0
#if USE_ASYNC_LDS
    if (k0 + 64 < K) {
      async_row32(la0, gA + k0 + 64);
      async_row32(lb0, gB + k0 + 64);
    }
#else
    {
      uint4 a0 = *(const uint4*)(gA + k0 + 32);
      uint4 a1 = *(const uint4*)(gA + k0 + 40);
      uint4 b0 = *(const uint4*)(gB + k0 + 32);
      uint4 b1 = *(const uint4*)(gB + k0 + 40);
      __syncthreads();
      *(uint4*)&lA[1][sr * 40 + sc] = a0;
      *(uint4*)&lA[1][sr * 40 + sc + 8] = a1;
      *(uint4*)&lB[1][sr * 40 + sc] = b0;
      *(uint4*)&lB[1][sr * 40 + sc + 8] = b1;
      __syncthreads();
    }
#endif
    {
      v16bf bf0 = frag_ld(lB[1], 40, wn, 0);
      v16bf bf1 = frag_ld(lB[1], 40, wn + 16, 0);
#pragma unroll
      for (int mt = 0; mt < 4; ++mt) {
        v16bf a = frag_ld(lA[1], 40, wm + mt * 16, 0);
        acc[mt][0] = wmma_bf16(a, bf0, acc[mt][0]);
        acc[mt][1] = wmma_bf16(a, bf1, acc[mt][1]);
      }
    }
#if USE_ASYNC_LDS
    wait_async0();
    __syncthreads();
#endif
  }

#pragma unroll
  for (int mt = 0; mt < 4; ++mt)
#pragma unroll
    for (int nt = 0; nt < 2; ++nt)
#pragma unroll
      for (int v = 0; v < 8; ++v) {
        int m = bm + wm + mt * 16 + v + ((lane >> 4) << 3);
        int n = bn + wn + nt * 16 + (lane & 15);
        float x = acc[mt][nt][v];
        if constexpr (MODE == 0) {
          float act = x / (1.f + __expf(-x));     // silu
          int which = n >> 11;
          int hd = n & 2047;
          int h = hd >> 7, d = hd & 127;
          int b = m >> 12, s = m & 4095;
          unsigned short* dst = (which == 0) ? qb : ((which == 1) ? kb : vb);
          dst[(((size_t)(b * HH + h) * SS) + s) * DD + d] = f2bf(act);
        } else if constexpr (MODE == 1) {
          gb[(size_t)m * HIDD + n] = f2bf(1.f / (1.f + __expf(-x)));  // sigmoid
        } else {
          of[(size_t)m * N + n] = x;
        }
      }
}

// ---- lightning attention: one workgroup per (b,h), 8 waves -----------------
__global__ __launch_bounds__(256) void lattn(
    const unsigned short* __restrict__ qb, const unsigned short* __restrict__ kb,
    const unsigned short* __restrict__ vb, const unsigned short* __restrict__ gate,
    const float* __restrict__ slopes, unsigned short* __restrict__ attn) {
  extern __shared__ char smem[];
  unsigned short* k_lds = (unsigned short*)smem;            // [j=256][d=128] s136
  unsigned short* v_lds = k_lds + 256 * 136;                // [e=128][j=256] s264
  float* kv_lds = (float*)(v_lds + 128 * 264);              // [e=128][d=128] s132
  unsigned short* s_all = (unsigned short*)(kv_lds + 128 * 132);
  float* dtab = (float*)(s_all + 8 * 16 * 264);             // exp(-s*t), t<512
  const int tid = threadIdx.x;
  const int lane = tid & 31;
  const int wid = tid >> 5;
  unsigned short* s_row = s_all + wid * (16 * 264);

  const int b = blockIdx.x >> 4;
  const int h = blockIdx.x & 15;
  const float sl = slopes[h];
  const float bdec = __expf(-sl * (float)CC);
  const size_t hbase = (size_t)blockIdx.x * SS * DD;

  for (int i = tid; i < 128 * 132; i += 256) kv_lds[i] = 0.f;
  for (int i = tid; i < 512; i += 256) dtab[i] = __expf(-sl * (float)i);

  for (int nc = 0; nc < NCC; ++nc) {
    __syncthreads();  // previous chunk fully retired
    const unsigned short* ksrc = kb + hbase + (size_t)nc * CC * DD;
    const unsigned short* vsrc = vb + hbase + (size_t)nc * CC * DD;
#pragma unroll
    for (int r = 0; r < 16; ++r) {
      int idx = r * 2048 + tid * 8;
      int row = idx >> 7, col = idx & 127;
#if USE_ASYNC_LDS
      unsigned lk = (unsigned)(unsigned long long)&k_lds[row * 136 + col];
      asm volatile("global_load_async_to_lds_b128 %0, %1, off"
                   :
                   : "v"(lk), "v"(ksrc + (size_t)row * 128 + col)
                   : "memory");
#else
      *(uint4*)&k_lds[row * 136 + col] =
          *(const uint4*)(ksrc + (size_t)row * 128 + col);
#endif
      // v staged transposed: v_lds[e][j]
      uint4 e4 = *(const uint4*)(vsrc + (size_t)row * 128 + col);
      unsigned short tmp[8];
      *(uint4*)tmp = e4;
#pragma unroll
      for (int u = 0; u < 8; ++u) v_lds[(col + u) * 264 + row] = tmp[u];
    }
#if USE_ASYNC_LDS
    wait_async0();
#endif
    __syncthreads();

    for (int half = 0; half < 2; ++half) {
      const int mt = wid + half * 8;
      const int i_base = mt * 16;
      const unsigned short* qrow = qb + hbase + (size_t)(nc * CC + i_base) * DD;
      v16bf qf[4];
#pragma unroll
      for (int kk = 0; kk < 4; ++kk) qf[kk] = frag_ld_g(qrow, kk * 32);

      // S = q @ k^T, causal decay (LDS table), restage bf16 per-wave
      for (int jt = 0; jt < 16; ++jt) {
        v8f sacc = vzero8();
#pragma unroll
        for (int kk = 0; kk < 4; ++kk)
          sacc = wmma_bf16(qf[kk], frag_ld(k_lds, 136, jt * 16, kk * 32), sacc);
#pragma unroll
        for (int v = 0; v < 8; ++v) {
          int ml = v + ((lane >> 4) << 3);
          int i = i_base + ml;
          int j = jt * 16 + (lane & 15);
          float dec = (i >= j) ? dtab[i - j] : 0.f;
          s_row[ml * 264 + j] = f2bf(sacc[v] * dec);
        }
      }

      // O = S_dec @ v (intra) ; O += (q*q_decay) @ kv (inter)
      v8f o[8];
#pragma unroll
      for (int et = 0; et < 8; ++et) o[et] = vzero8();
      for (int kk = 0; kk < 8; ++kk) {
        v16bf af = frag_ld(s_row, 264, 0, kk * 32);
#pragma unroll
        for (int et = 0; et < 8; ++et)
          o[et] = wmma_bf16(af, frag_ld(v_lds, 264, et * 16, kk * 32), o[et]);
      }
      float qd = dtab[i_base + (lane & 15) + 1];
#pragma unroll
      for (int kk = 0; kk < 4; ++kk) {
        v16bf aq = frag_scale(qf[kk], qd);
#pragma unroll
        for (int et = 0; et < 8; ++et)
          o[et] = wmma_bf16(aq, frag_ld_kv(kv_lds, et * 16, kk * 32), o[et]);
      }
      // epilogue: fuse output gate, store gated attn bf16 [B*S][H*D]
#pragma unroll
      for (int et = 0; et < 8; ++et)
#pragma unroll
        for (int v = 0; v < 8; ++v) {
          int ml = v + ((lane >> 4) << 3);
          size_t row = (size_t)b * SS + nc * CC + i_base + ml;
          size_t idx = row * HIDD + h * DD + et * 16 + (lane & 15);
          attn[idx] = f2bf(o[et][v] * bf2f(gate[idx]));
        }
    }
    __syncthreads();  // all inter-chunk reads of kv done before update

    // kv = bdec*kv + (k*k_decay)^T @ v ; wave owns d-rows [wid*16, wid*16+16)
    v8f u[8];
#pragma unroll
    for (int et = 0; et < 8; ++et) u[et] = vzero8();
    for (int kk = 0; kk < 8; ++kk) {
      v16bf af = frag_ld_kt(k_lds, dtab, wid * 16, kk * 32);
#pragma unroll
      for (int et = 0; et < 8; ++et)
        u[et] = wmma_bf16(af, frag_ld(v_lds, 264, et * 16, kk * 32), u[et]);
    }
#pragma unroll
    for (int et = 0; et < 8; ++et)
#pragma unroll
      for (int v = 0; v < 8; ++v) {
        int d = wid * 16 + v + ((lane >> 4) << 3);
        int e = et * 16 + (lane & 15);
        float* pk = &kv_lds[e * 132 + d];
        *pk = *pk * bdec + u[et][v];
      }
  }
}

extern "C" void kernel_launch(void* const* d_in, const int* in_sizes, int n_in,
                              void* d_out, int out_size, void* d_ws, size_t ws_size,
                              hipStream_t stream) {
  const float* x      = (const float*)d_in[0];
  const float* w_qkv  = (const float*)d_in[1];
  const float* w_gate = (const float*)d_in[2];
  const float* w_out  = (const float*)d_in[3];
  const float* slopes = (const float*)d_in[4];

  const size_t NE = (size_t)BB * SS * HH * DD;       // 16,777,216
  unsigned short* qb = (unsigned short*)d_ws;
  unsigned short* kb = qb + NE;
  unsigned short* vb = kb + NE;
  unsigned short* gb = vb + NE;
  unsigned short* ab = gb + NE;
  unsigned short* xh = ab + NE;                      // x in bf16
  unsigned short* wqkvT  = xh + NE;                  // [6144][2048]
  unsigned short* wgateT = wqkvT + (size_t)3 * HIDD * HIDD;
  unsigned short* woutT  = wgateT + (size_t)HIDD * HIDD;
  float* out = (float*)d_out;

  const int M = BB * SS;  // 8192
  dim3 blk(256);

  // one-shot bf16 conversion / weight transpose
  cvt_bf16<<<dim3(NE / 1024), blk, 0, stream>>>(x, xh);
  cvt_wT<<<dim3((3 * HIDD) / 32, HIDD / 32), blk, 0, stream>>>(w_qkv, wqkvT, HIDD, 3 * HIDD);
  cvt_wT<<<dim3(HIDD / 32, HIDD / 32), blk, 0, stream>>>(w_gate, wgateT, HIDD, HIDD);
  cvt_wT<<<dim3(HIDD / 32, HIDD / 32), blk, 0, stream>>>(w_out, woutT, HIDD, HIDD);

  gemm_k<0><<<dim3((3 * HIDD) / 128, M / 128), blk, 0, stream>>>(
      xh, wqkvT, M, 3 * HIDD, HIDD, qb, kb, vb, nullptr, nullptr);
  gemm_k<1><<<dim3(HIDD / 128, M / 128), blk, 0, stream>>>(
      xh, wgateT, M, HIDD, HIDD, nullptr, nullptr, nullptr, gb, nullptr);

  size_t shbytes = (size_t)(256 * 136 + 128 * 264) * 2 + (size_t)128 * 132 * 4 +
                   (size_t)8 * 16 * 264 * 2 + 512 * 4;   // 274,432 B < 320 KB
  lattn<<<dim3(BB * HH), blk, shbytes, stream>>>(qb, kb, vb, gb, slopes, ab);

  gemm_k<2><<<dim3(HIDD / 128, M / 128), blk, 0, stream>>>(
      ab, woutT, M, HIDD, HIDD, nullptr, nullptr, nullptr, nullptr, out);
}
